// LowRankRotatedSpaceIntervention_14680198218307
// MI455X (gfx1250) — matrix operand order repaired and checked by reading, not compiled, not run on hardware
//
#include <hip/hip_runtime.h>

// Problem constants (from reference): B=16384, D=4096, R=64, N_PART=8, P=8, N_SEL=4
#define B_TOK 16384
#define DDIM  4096
#define RDIM  64

typedef __attribute__((ext_vector_type(16))) __bf16       v16bf;
typedef __attribute__((ext_vector_type(8)))  float        v8f;
typedef __attribute__((ext_vector_type(4)))  float        v4f;
typedef __attribute__((ext_vector_type(8)))  unsigned int v8u;
typedef __attribute__((ext_vector_type(4)))  unsigned int v4u;

__device__ __forceinline__ unsigned short bf16bits(float f) {
  return __builtin_bit_cast(unsigned short, (__bf16)f);   // native cvt, RNE
}

// ---------------------------------------------------------------------------
// Prep: swizzle W (fp32 [4096][64]) into bf16 WMMA B-operand register images.
//   wb: W   as B-operand tiles, tile t = kt*4+nt   (kt: K/32, nt: N/16), phase A
//   wt: W^T as B-operand tiles, tile t = nt2*2+kt2 (nt2: D/16, kt2: R/32), phase B
// Per ISA 16-bit B layout (32x16): lane l holds column n=l&15; element e (=2i+p)
// holds K = (l>>4)*8 + (e&7) + (e>>3)*16. Each tile = 32 lanes x 16 bf16 = 1024 bytes.
// ---------------------------------------------------------------------------
__global__ void __launch_bounds__(256) prep_w_kernel(const float* __restrict__ W,
                                                     unsigned short* __restrict__ wb,
                                                     unsigned short* __restrict__ wt) {
  const int gid  = blockIdx.x * 256 + threadIdx.x;   // 0..32767
  const int id   = gid & 16383;
  const int t    = id >> 5;
  const int lane = id & 31;
  const int h16  = (lane >> 4) * 16;
  const int nl   = lane & 15;
  if (gid < 16384) {
    const int kt = t >> 2, nt = t & 3;               // K-tile of 32, N-tile of 16
    unsigned short* dst = wb + (size_t)t * 512 + lane * 16;
#pragma unroll
    for (int e = 0; e < 16; ++e) {
      const int k = kt * 32 + (h16 >> 1) + (e & 7) + ((e >> 3) << 4);
      const int n = nt * 16 + nl;
      dst[e] = bf16bits(W[(size_t)k * RDIM + n]);
    }
  } else {
    const int nt2 = t >> 1, kt2 = t & 1;             // D-tile of 16, R-tile of 32
    unsigned short* dst = wt + (size_t)t * 512 + lane * 16;
#pragma unroll
    for (int e = 0; e < 16; ++e) {
      const int r = kt2 * 32 + (h16 >> 1) + (e & 7) + ((e >> 3) << 4);
      const int d = nt2 * 16 + nl;
      dst[e] = bf16bits(W[(size_t)d * RDIM + r]);    // W^T[r][d]
    }
  }
}

// ---------------------------------------------------------------------------
// Fused main kernel: one wave (32 threads) handles 16 rows.
//   Phase A: diff[16x64] = (source - base)[16x4096] @ W[4096x64]  (128 K-steps)
//   Mask+store diff to LDS as bf16; reload in A-operand layout.
//   Phase B: out[16x4096] = base + maskeddiff[16x64] @ W^T[64x4096] (256 N-tiles)
// Cache policy: source = NT load (single use), base = RT (reused in phase B via
// L2), out = NT store, W tiles = RT (shared by all 1024 workgroups, L2-resident).
// ---------------------------------------------------------------------------
__global__ void __launch_bounds__(32) lowrank_main_kernel(
    const float* __restrict__ base, const float* __restrict__ source,
    const int* __restrict__ subspaces,
    const unsigned short* __restrict__ wb, const unsigned short* __restrict__ wt,
    float* __restrict__ out) {
  __shared__ __align__(16) unsigned short sdiff[16 * 64];
  __shared__ unsigned long long smask[16];

  const int lane  = threadIdx.x;          // 0..31
  const int mbase = blockIdx.x * 16;
  const int h     = lane >> 4;            // half-wave id
  const int m     = lane & 15;            // row within tile / column within tile

  // Per-row selection masks over R=64 rotated dims (lanes 0..15).
  if (lane < 16) {
    unsigned long long mk = 0ull;
    const int* sp = subspaces + (size_t)(mbase + lane) * 4;
#pragma unroll
    for (int s = 0; s < 4; ++s) {
      const int p = sp[s] & 7;
      mk |= 0xFFull << (p * 8);
    }
    smask[lane] = mk;
  }

  // -------- Phase A: diff = (source - base) @ W --------
  v8f acc[4] = {};   // four 16x16 f32 accumulators: N = 0..63

  const float* srow = source + (size_t)(mbase + m) * DDIM;
  const float* brow = base   + (size_t)(mbase + m) * DDIM;

  for (int kt = 0; kt < 128; ++kt) {
    // A-operand (16x32 bf16): lane half h covers K in
    // {h*8..h*8+7} U {16+h*8..16+h*8+7} within this 32-wide K slab.
    const int k0 = kt * 32 + h * 8;
    const v4f s0 = __builtin_nontemporal_load((const v4f*)(srow + k0));
    const v4f s1 = __builtin_nontemporal_load((const v4f*)(srow + k0 + 4));
    const v4f s2 = __builtin_nontemporal_load((const v4f*)(srow + k0 + 16));
    const v4f s3 = __builtin_nontemporal_load((const v4f*)(srow + k0 + 20));
    const v4f b0 = *(const v4f*)(brow + k0);
    const v4f b1 = *(const v4f*)(brow + k0 + 4);
    const v4f b2 = *(const v4f*)(brow + k0 + 16);
    const v4f b3 = *(const v4f*)(brow + k0 + 20);

    v16bf av;
#pragma unroll
    for (int e = 0; e < 4; ++e) {
      av[e]      = (__bf16)(s0[e] - b0[e]);   // K = k0   .. k0+3
      av[4 + e]  = (__bf16)(s1[e] - b1[e]);   // K = k0+4 .. k0+7
      av[8 + e]  = (__bf16)(s2[e] - b2[e]);   // K = k0+16.. k0+19
      av[12 + e] = (__bf16)(s3[e] - b3[e]);   // K = k0+20.. k0+23
    }

#pragma unroll
    for (int nt = 0; nt < 4; ++nt) {
      const v8u braw = ((const v8u*)(wb + (size_t)(kt * 4 + nt) * 512))[lane];
      const v16bf bv = __builtin_bit_cast(v16bf, braw);
      acc[nt] = __builtin_amdgcn_wmma_f32_16x16x32_bf16(
          false, av, false, bv, (short)0, acc[nt], false, false);
    }
  }

  // Masks were written via LDS by lanes 0..15; single wave + in-order LDS, but
  // wait explicitly before the cross-lane read.
  asm volatile("s_wait_dscnt 0x0" ::: "memory");

  // Apply mask, convert to bf16, stage diff tile in LDS (C layout -> row major).
  {
    const int rbase = h * 8;                 // C layout: lanes>=16 hold rows 8..15
#pragma unroll
    for (int nt = 0; nt < 4; ++nt) {
      const int r = nt * 16 + m;             // rotated-dim index (column of diff)
#pragma unroll
      for (int j = 0; j < 8; ++j) {
        const int row = rbase + j;
        const unsigned long long mk = smask[row];
        const float v = ((mk >> r) & 1ull) ? acc[nt][j] : 0.0f;
        sdiff[row * 64 + r] = bf16bits(v);
      }
    }
  }
  asm volatile("s_wait_dscnt 0x0" ::: "memory");

  // -------- Phase B: out = base + maskeddiff @ W^T --------
  // Reload diff as two 16x32 bf16 A-operand tiles (K = rotated dim).
  v16bf a2[2];
#pragma unroll
  for (int kk = 0; kk < 2; ++kk) {
    const v4u lo = *(const v4u*)&sdiff[m * 64 + kk * 32 + h * 8];
    const v4u hi = *(const v4u*)&sdiff[m * 64 + kk * 32 + 16 + h * 8];
    v8u comb;
    comb[0] = lo[0]; comb[1] = lo[1]; comb[2] = lo[2]; comb[3] = lo[3];
    comb[4] = hi[0]; comb[5] = hi[1]; comb[6] = hi[2]; comb[7] = hi[3];
    a2[kk] = __builtin_bit_cast(v16bf, comb);
  }

  const size_t rowoff = (size_t)(mbase + h * 8) * DDIM;
  for (int nt2 = 0; nt2 < 256; ++nt2) {
    v8f c = {};
#pragma unroll
    for (int kk = 0; kk < 2; ++kk) {
      const v8u braw = ((const v8u*)(wt + (size_t)(nt2 * 2 + kk) * 512))[lane];
      const v16bf bv = __builtin_bit_cast(v16bf, braw);
      c = __builtin_amdgcn_wmma_f32_16x16x32_bf16(
          false, a2[kk], false, bv, (short)0, c, false, false);
    }
    const int d = nt2 * 16 + m;              // output column
#pragma unroll
    for (int j = 0; j < 8; ++j) {
      const size_t off = rowoff + (size_t)j * DDIM + d;
      __builtin_nontemporal_store(base[off] + c[j], &out[off]);  // fused add
    }
  }
}

extern "C" void kernel_launch(void* const* d_in, const int* in_sizes, int n_in,
                              void* d_out, int out_size, void* d_ws, size_t ws_size,
                              hipStream_t stream) {
  const float* base      = (const float*)d_in[0];
  const float* source    = (const float*)d_in[1];
  const float* W         = (const float*)d_in[2];
  const int*   subspaces = (const int*)d_in[3];
  float*       out       = (float*)d_out;

  // Workspace: wb = 4096*64 bf16 (512 KB), wt = 64*4096 bf16 (512 KB).
  unsigned short* wb = (unsigned short*)d_ws;
  unsigned short* wt = wb + (size_t)DDIM * RDIM;

  prep_w_kernel<<<128, 256, 0, stream>>>(W, wb, wt);
  lowrank_main_kernel<<<B_TOK / 16, 32, 0, stream>>>(base, source, subspaces,
                                                     wb, wt, out);
}
